// EG3DDecoder_43542378447417
// MI455X (gfx1250) — matrix operand order repaired and checked by reading, not compiled, not run on hardware
//
#include <hip/hip_runtime.h>
#include <hip/hip_bf16.h>

typedef __attribute__((ext_vector_type(16))) _Float16 v16h;
typedef __attribute__((ext_vector_type(8)))  float    v8f;

#define VOL  64
#define CS   48
#define FS   48
#define FEAT 32
#define NRAY (4 * VOL * VOL)              // 16384
#define NPTS (NRAY * CS)                  // 786432 (same for fine)
#define RES  256
#define PLSZ (RES * RES)                  // 65536 floats per channel plane

// ---------------------------------------------------------------------------
// Kernel 1: per-ray origins/dirs.  o = R@(lx,ly,0)+t,  d = -R[:,2]
// ---------------------------------------------------------------------------
__global__ __launch_bounds__(256) void ray_setup(const float* __restrict__ c2w,
                                                 float* __restrict__ o_ws,
                                                 float* __restrict__ d_ws) {
  int rr = blockIdx.x * blockDim.x + threadIdx.x;
  if (rr >= NRAY) return;
  int b = rr >> 12, n = rr & 4095;
  int iy = n >> 6, ix = n & 63;
  const float scale = tanf(0.10471975511965977f); // 12deg * pi/360
  float lx = (-1.f + 2.f * ix / 63.f) * scale;
  float ly = (-1.f + 2.f * iy / 63.f) * scale;
  const float* M = c2w + b * 12; // (3,4) row-major
#pragma unroll
  for (int i = 0; i < 3; ++i) {
    o_ws[rr * 3 + i] = M[i * 4 + 0] * lx + M[i * 4 + 1] * ly + M[i * 4 + 3];
    d_ws[rr * 3 + i] = -M[i * 4 + 2];
  }
}

// ---------------------------------------------------------------------------
// Kernel 2/4: triplane sample + 2-layer MLP via WMMA.
// Block = 128 threads (4 waves). Wave does 4 groups of 16 points.
// A-matrix (16x32 f16):  lane<16: M=lane, K in {0..7,16..23}; lane>=16: K in {8..15,24..31}
// B-matrix (32x16 f16):  lane<16: N=lane, K in {0..7,16..23}; lane>=16: K in {8..15,24..31}
// C/D (16x16 f32):       elem r -> M = r + 8*(lane>=16),       N = lane&15
// ---------------------------------------------------------------------------
__global__ __launch_bounds__(128) void decode_kernel(
    const float* __restrict__ tp, const float* __restrict__ W1,
    const float* __restrict__ b1, const float* __restrict__ W2,
    const float* __restrict__ b2, const float* __restrict__ o_ws,
    const float* __restrict__ d_ws, const float* __restrict__ zbuf, // null => coarse linspace
    _Float16* __restrict__ feat_out, float* __restrict__ sig_out) {
  const int lane = threadIdx.x & 31;
  const int wave = threadIdx.x >> 5;
  const int hi   = lane >> 4;   // lane-half
  const int ml   = lane & 15;   // M (for A rows) or N (for B/C cols)

  __shared__ _Float16 lds_h[4][16][66]; // [wave][M][64 cols] (+pad)

  // ---- hoisted weight tiles in matrix-core layouts ----
  v16h w1t[4];
#pragma unroll
  for (int t = 0; t < 4; ++t)
#pragma unroll
    for (int j = 0; j < 16; ++j) {
      int K = j + ((j >= 8) ? 8 : 0) + hi * 8; // 0..31
      w1t[t][j] = (_Float16)W1[K * 64 + t * 16 + ml];
    }
  float bias1[4];
#pragma unroll
  for (int t = 0; t < 4; ++t) bias1[t] = b1[t * 16 + ml];

  v16h w2t[3][2];
#pragma unroll
  for (int t = 0; t < 3; ++t)
#pragma unroll
    for (int ck = 0; ck < 2; ++ck)
#pragma unroll
      for (int j = 0; j < 16; ++j) {
        int K = j + ((j >= 8) ? 8 : 0) + hi * 8 + ck * 32; // 0..63
        int col = t * 16 + ml;
        w2t[t][ck][j] = (col < 33) ? (_Float16)W2[K * 33 + col] : (_Float16)0.f;
      }
  float bias2[3];
#pragma unroll
  for (int t = 0; t < 3; ++t) {
    int col = t * 16 + ml;
    bias2[t] = (col < 33) ? b2[col] : 0.f;
  }
  const v8f zacc = {0.f, 0.f, 0.f, 0.f, 0.f, 0.f, 0.f, 0.f};

  for (int g = 0; g < 4; ++g) {
    const long pbase = (long)blockIdx.x * 256 + wave * 64 + g * 16;
    const long pm = pbase + ml;          // this lane's point (2 lanes per point)
    const int  r  = (int)(pm / CS);
    const int  s  = (int)(pm % CS);
    const int  b  = r >> 12;
    const float z = zbuf ? zbuf[pm] : s * (2.0f / 47.0f);

    float px = o_ws[r * 3 + 0] + d_ws[r * 3 + 0] * z;
    float py = o_ws[r * 3 + 1] + d_ws[r * 3 + 1] * z;
    float pz = o_ws[r * 3 + 2] + d_ws[r * 3 + 2] * z;
    float x  = fminf(fmaxf(px * 0.4f, -1.f), 1.f);          // /2.5
    float y  = fminf(fmaxf(py * 0.5f, -1.f), 1.f);          // /2.0
    float zz = fminf(fmaxf((pz + 2.f) * 0.4f, -1.f), 1.f);  // (+2)/2.5

    float us[3] = {x, x, y};
    float vs[3] = {y, zz, zz};

    float acc[16];
#pragma unroll
    for (int j = 0; j < 16; ++j) acc[j] = 0.f;

    for (int pl = 0; pl < 3; ++pl) {
      float gx = (us[pl] + 1.f) * 127.5f;
      float gy = (vs[pl] + 1.f) * 127.5f;
      float x0f = fminf(fmaxf(floorf(gx), 0.f), 254.f);
      float y0f = fminf(fmaxf(floorf(gy), 0.f), 254.f);
      float wx = gx - x0f, wy = gy - y0f;
      const float* base =
          tp + ((size_t)(b * 3 + pl) * FEAT) * PLSZ + (size_t)((int)y0f) * RES + (int)x0f;
#pragma unroll
      for (int j = 0; j < 16; ++j) {
        int c = j + ((j >= 8) ? 8 : 0) + hi * 8; // A-layout channel for this lane
        const float* q = base + (size_t)c * PLSZ;
        float p00 = q[0], p01 = q[1], p10 = q[RES], p11 = q[RES + 1];
        acc[j] += (p00 * (1.f - wx) + p01 * wx) * (1.f - wy) +
                  (p10 * (1.f - wx) + p11 * wx) * wy;
      }
    }
    v16h A;
#pragma unroll
    for (int j = 0; j < 16; ++j) A[j] = (_Float16)acc[j];

    // ---- layer 1: h(16x64) = A(16x32) @ W1(32x64), 4 column tiles ----
    v8f h[4];
#pragma unroll
    for (int t = 0; t < 4; ++t) {
      h[t] = __builtin_amdgcn_wmma_f32_16x16x32_f16(false, A, false, w1t[t],
                                                    (short)0, zacc, false, false);
#pragma unroll
      for (int rr2 = 0; rr2 < 8; ++rr2)
        h[t][rr2] = fmaxf(h[t][rr2] + bias1[t], 0.f);
    }

    // ---- re-swizzle C-layout -> A-layout via per-wave LDS (in-order DS) ----
#pragma unroll
    for (int t = 0; t < 4; ++t)
#pragma unroll
      for (int rr2 = 0; rr2 < 8; ++rr2)
        lds_h[wave][rr2 + hi * 8][t * 16 + ml] = (_Float16)h[t][rr2];

    v16h a2[2];
#pragma unroll
    for (int ck = 0; ck < 2; ++ck)
#pragma unroll
      for (int j = 0; j < 16; ++j) {
        int K = j + ((j >= 8) ? 8 : 0) + hi * 8 + ck * 32;
        a2[ck][j] = lds_h[wave][ml][K];
      }

    // ---- layer 2: out(16x33) = h(16x64) @ W2(64x33), chained over K ----
#pragma unroll
    for (int t = 0; t < 3; ++t) {
      v8f oa = __builtin_amdgcn_wmma_f32_16x16x32_f16(false, a2[0], false, w2t[t][0],
                                                      (short)0, zacc, false, false);
      oa = __builtin_amdgcn_wmma_f32_16x16x32_f16(false, a2[1], false, w2t[t][1],
                                                  (short)0, oa, false, false);
      int col = t * 16 + ml;
#pragma unroll
      for (int rr2 = 0; rr2 < 8; ++rr2) {
        long p = pbase + rr2 + hi * 8;
        float val = oa[rr2] + bias2[t];
        if (col < FEAT)       feat_out[p * FEAT + col] = (_Float16)val;
        else if (col == FEAT) sig_out[p] = val;
      }
    }
  }
}

// ---------------------------------------------------------------------------
// Kernel 3: coarse composite weights + inverse-CDF fine-z (thread per ray)
// ---------------------------------------------------------------------------
__global__ __launch_bounds__(256) void coarse_pdf(const float* __restrict__ sig_c,
                                                  float* __restrict__ fine_z) {
  int rr = blockIdx.x * blockDim.x + threadIdx.x;
  if (rr >= NRAY) return;
  const float dz = 2.0f / 47.0f;
  float w[CS];
  float T = 1.f;
  for (int s = 0; s < CS; ++s) {
    float sg = fmaxf(sig_c[(size_t)rr * CS + s], 0.f);
    float delta = (s < CS - 1) ? dz : 1e10f;
    float a = 1.f - __expf(-delta * sg);
    w[s] = a * T + 1e-5f;
    T *= (1.f - a + 1e-10f);
  }
  // bins zmid[i] = (i+0.5)*dz, i=0..46 ; weights used: w[1..46]
  float sum = 0.f;
  for (int i = 0; i < 46; ++i) sum += w[i + 1];
  float cdf[47];
  cdf[0] = 0.f;
  float c = 0.f;
  for (int i = 0; i < 46; ++i) { c += w[i + 1] / sum; cdf[i + 1] = c; }
  for (int k = 0; k < FS; ++k) {
    float u = (float)k / 47.0f;
    int ind = 0;
    for (int i = 0; i < 47; ++i) ind += (cdf[i] <= u) ? 1 : 0; // searchsorted 'right'
    int below = max(ind - 1, 0);
    int above = min(ind, 46);
    float cb = cdf[below], ca = cdf[above];
    float bb = (below + 0.5f) * dz, ba = (above + 0.5f) * dz;
    float denom = (ca - cb < 1e-5f) ? 1.f : (ca - cb);
    fine_z[(size_t)rr * FS + k] = bb + (u - cb) / denom * (ba - bb);
  }
}

// ---------------------------------------------------------------------------
// Kernel 5: stable 3-way merge + final composite (wave per ray, lane=channel)
// ---------------------------------------------------------------------------
__global__ __launch_bounds__(128) void final_composite(
    const float* __restrict__ fine_z, const float* __restrict__ sig_c,
    const float* __restrict__ sig_f, const _Float16* __restrict__ feat_c,
    const _Float16* __restrict__ feat_f, const float* __restrict__ ftex,
    const float* __restrict__ falpha, const float* __restrict__ fdepth,
    float* __restrict__ out) {
  int gw = (blockIdx.x * blockDim.x + threadIdx.x) >> 5;
  int lane = threadIdx.x & 31;
  if (gw >= NRAY) return;
  int rr = gw, b = rr >> 12, n = rr & 4095;
  const float dz = 2.0f / 47.0f;
  const float INF = 3.4e38f;
  float zface = fdepth[b * 4096 + n];
  float sface = falpha[b * 4096 + n];
  float fface = ftex[((size_t)b * FEAT + lane) * 4096 + n];

  int i = 0, j = 0;
  bool faceleft = true;
  float T = 1.f, acc = 0.f, depth = 0.f;
  float zp = 0.f, sp = 0.f, fp = 0.f;
  int wasface = 0, havep = 0;
  for (int step = 0; step < FS + CS + 1; ++step) {
    float zf = (i < FS) ? fine_z[(size_t)rr * FS + i] : INF;
    float zc = (j < CS) ? j * dz : INF;
    float zd = faceleft ? zface : INF;
    float z, sg, ft;
    int isf = 0;
    if (zf <= zc && zf <= zd) {           // fine first on ties (stable)
      z = zf; sg = sig_f[(size_t)rr * FS + i];
      ft = (float)feat_f[((size_t)rr * FS + i) * FEAT + lane]; ++i;
    } else if (zc <= zd) {                // coarse before face on ties
      z = zc; sg = sig_c[(size_t)rr * CS + j];
      ft = (float)feat_c[((size_t)rr * CS + j) * FEAT + lane]; ++j;
    } else {
      z = zd; sg = sface; ft = fface; isf = 1; faceleft = false;
    }
    if (havep) {
      float delta = z - zp;
      float a = wasface ? sp : 1.f - __expf(-delta * fmaxf(sp, 0.f));
      float wgt = a * T;
      acc += wgt * fp; depth += wgt * zp;
      T *= (1.f - a + 1e-10f);
    }
    zp = z; sp = sg; fp = ft; wasface = isf; havep = 1;
  }
  { // last sample: delta = 1e10
    float a = wasface ? sp : 1.f - __expf(-1e10f * fmaxf(sp, 0.f));
    float wgt = a * T;
    acc += wgt * fp; depth += wgt * zp;
  }
  out[(size_t)rr * FEAT + lane] = acc;
  if (lane == 0) out[(size_t)NRAY * FEAT + rr] = depth;
}

// ---------------------------------------------------------------------------
extern "C" void kernel_launch(void* const* d_in, const int* in_sizes, int n_in,
                              void* d_out, int out_size, void* d_ws, size_t ws_size,
                              hipStream_t stream) {
  const float* tp     = (const float*)d_in[0];
  const float* W1     = (const float*)d_in[1];
  const float* b1     = (const float*)d_in[2];
  const float* W2     = (const float*)d_in[3];
  const float* b2     = (const float*)d_in[4];
  const float* c2w    = (const float*)d_in[5];
  const float* ftex   = (const float*)d_in[6];
  const float* falpha = (const float*)d_in[7];
  const float* fdepth = (const float*)d_in[8];
  float* out = (float*)d_out;

  float* o_ws   = (float*)d_ws;
  float* d_dir  = o_ws + (size_t)NRAY * 3;
  float* sig_c  = d_dir + (size_t)NRAY * 3;
  float* sig_f  = sig_c + (size_t)NPTS;
  float* fine_z = sig_f + (size_t)NPTS;
  _Float16* feat_c = (_Float16*)(fine_z + (size_t)NPTS);
  _Float16* feat_f = feat_c + (size_t)NPTS * FEAT;

  ray_setup<<<NRAY / 256, 256, 0, stream>>>(c2w, o_ws, d_dir);
  decode_kernel<<<NPTS / 256, 128, 0, stream>>>(tp, W1, b1, W2, b2, o_ws, d_dir,
                                                nullptr, feat_c, sig_c);
  coarse_pdf<<<NRAY / 256, 256, 0, stream>>>(sig_c, fine_z);
  decode_kernel<<<NPTS / 256, 128, 0, stream>>>(tp, W1, b1, W2, b2, o_ws, d_dir,
                                                fine_z, feat_f, sig_f);
  final_composite<<<NRAY * 32 / 128, 128, 0, stream>>>(fine_z, sig_c, sig_f, feat_c,
                                                       feat_f, ftex, falpha, fdepth, out);
}
